// LSTMDecoder_23416161698415
// MI455X (gfx1250) — matrix operand order repaired and checked by reading, not compiled, not run on hardware
//
#include <hip/hip_runtime.h>
#include <hip/hip_bf16.h>
#include <math.h>

typedef _Float16 v16h __attribute__((ext_vector_type(16)));
typedef _Float16 v8h  __attribute__((ext_vector_type(8)));
typedef float    v8f  __attribute__((ext_vector_type(8)));

#define T_LEN 512
#define BATCH 1024

__device__ __forceinline__ float fast_tanh(float x) {
#if __has_builtin(__builtin_amdgcn_tanhf)
    return __builtin_amdgcn_tanhf(x);
#else
    return tanhf(x);
#endif
}
__device__ __forceinline__ float fast_sig(float x) {
    return 0.5f * fast_tanh(0.5f * x) + 0.5f;
}
__device__ __forceinline__ v8f zero8() {
    return (v8f){0.f, 0.f, 0.f, 0.f, 0.f, 0.f, 0.f, 0.f};
}
__device__ __forceinline__ v8f wmma16(v16h a, v16h b, v8f c) {
    // D = A(16x32,f16) * B(32x16,f16) + C(16x16,f32)
    return __builtin_amdgcn_wmma_f32_16x16x32_f16(false, a, false, b, (short)0, c,
                                                  false, false);
}

// A-fragment (16x32 f16): lane L: row = L&15; halfs 0..7 = K kb..kb+7,
// halfs 8..15 = K kb+16..kb+23, kb = 32*chunk + (L<16 ? 0 : 8).
__device__ __forceinline__ v16h load_afrag(const _Float16* p, int lane, int c) {
    int row = lane & 15;
    int kb  = c * 32 + ((lane & 16) ? 8 : 0);
    v8h lo = *(const v8h*)(p + row * 128 + kb);
    v8h hi = *(const v8h*)(p + row * 128 + kb + 16);
    return __builtin_shufflevector(lo, hi, 0, 1, 2, 3, 4, 5, 6, 7,
                                   8, 9, 10, 11, 12, 13, 14, 15);
}

// B-fragment stored pre-formatted: 16 contiguous halfs per lane.
__device__ __forceinline__ v16h load_bfrag(const _Float16* p, int lane, int s, int n) {
    const _Float16* q = p + ((s * 17 + n) * 512) + lane * 16;
    v8h lo = *(const v8h*)q;
    v8h hi = *(const v8h*)(q + 8);
    return __builtin_shufflevector(lo, hi, 0, 1, 2, 3, 4, 5, 6, 7,
                                   8, 9, 10, 11, 12, 13, 14, 15);
}

// Layer-1 BiLSTM + layer-2 input projections.
// One wave (32 threads) handles 16 batch rows for all 512 steps, both directions.
// A-panel K layout: k0..49 = h, k50 = 1.0 (bias lane), k51..63 = 0, k64..127 = x.
// N layout: cols 0..255 = 4 gates x 64 (j<50 valid), cols 256..271 = layer-2 proj tile.
__global__ __launch_bounds__(32) void lstm1_kernel(
    const float* __restrict__ x,
    const float* __restrict__ Wih_f, const float* __restrict__ Whh_f,
    const float* __restrict__ bih_f, const float* __restrict__ bhh_f,
    const float* __restrict__ Wih_b, const float* __restrict__ Whh_b,
    const float* __restrict__ bih_b, const float* __restrict__ bhh_b,
    const float* __restrict__ Wih2f, const float* __restrict__ bih2f,
    const float* __restrict__ bhh2f,
    const float* __restrict__ Wih2b, const float* __restrict__ bih2b,
    const float* __restrict__ bhh2b,
    _Float16* __restrict__ ws)  // [T][BATCH][16] f16
{
    __shared__ __attribute__((aligned(16))) _Float16 lds_w[2 * 17 * 512];  // 34 KB
    __shared__ __attribute__((aligned(16))) _Float16 lds_a[16 * 128];      // 4 KB

    const int lane  = threadIdx.x;
    const int ln    = lane & 15;
    const int rbase = (lane & 16) ? 8 : 0;
    const int row0  = blockIdx.x * 16;

    for (int dir = 0; dir < 2; ++dir) {
        const float* Wih = dir ? Wih_b : Wih_f;
        const float* Whh = dir ? Whh_b : Whh_f;
        const float* bih = dir ? bih_b : bih_f;
        const float* bhh = dir ? bhh_b : bhh_f;

        // ---- init A panel ----
        for (int idx = lane; idx < 16 * 128; idx += 32) {
            int row = idx >> 7, kk = idx & 127;
            float v = 0.f;
            if (kk >= 64)      v = x[(row0 + row) * 64 + (kk - 64)];
            else if (kk == 50) v = 1.f;
            lds_a[idx] = (_Float16)v;
        }

        // ---- phase X: B fragments for x-part (global k 64..127) ----
        for (int idx = lane; idx < 2 * 17 * 512; idx += 32) {
            int h = idx & 15, l = (idx >> 4) & 31, fn = idx >> 9;
            int s = fn / 17, nt = fn % 17;
            int col = nt * 16 + (l & 15);
            int k = s * 32 + ((l & 16) ? 8 : 0) + (h < 8 ? h : h + 8);  // 0..63 = x dim
            float val = 0.f;
            if (col < 256) {
                int g = col >> 6, j = col & 63;
                if (j < 50) val = Wih[(g * 50 + j) * 64 + k];
            }
            lds_w[idx] = (_Float16)val;
        }
        __syncthreads();

        // ---- precompute constant input contribution xg via WMMA ----
        v16h a2 = load_afrag(lds_a, lane, 2);
        v16h a3 = load_afrag(lds_a, lane, 3);
        v8f xg[16];
#pragma unroll
        for (int n = 0; n < 16; ++n) {
            v8f t0 = wmma16(a2, load_bfrag(lds_w, lane, 0, n), zero8());
            xg[n]  = wmma16(a3, load_bfrag(lds_w, lane, 1, n), t0);
        }
        __syncthreads();

        // ---- phase H: B fragments for h-part (global k 0..63) + proj tile ----
        for (int idx = lane; idx < 2 * 17 * 512; idx += 32) {
            int h = idx & 15, l = (idx >> 4) & 31, fn = idx >> 9;
            int s = fn / 17, nt = fn % 17;
            int col = nt * 16 + (l & 15);
            int k = s * 32 + ((l & 16) ? 8 : 0) + (h < 8 ? h : h + 8);  // 0..49 = h, 50 = bias
            float val = 0.f;
            if (col < 256) {
                int g = col >> 6, j = col & 63;
                if (j < 50) {
                    int r = g * 50 + j;
                    if (k < 50)       val = Whh[r * 50 + k];
                    else if (k == 50) val = bih[r] + bhh[r];
                }
            } else {
                int m = col - 256;
                if (m < 8) {
                    int mm = m & 3;
                    if (k < 50) {
                        const float* W2 = (m < 4) ? Wih2f : Wih2b;
                        val = W2[mm * 100 + k + (dir ? 50 : 0)];
                    } else if (k == 50 && dir == 0) {
                        val = (m < 4) ? (bih2f[mm] + bhh2f[mm])
                                      : (bih2b[mm] + bhh2b[mm]);
                    }
                }
            }
            lds_w[idx] = (_Float16)val;
        }
        __syncthreads();

        v16h a0 = load_afrag(lds_a, lane, 0);
        v16h a1 = load_afrag(lds_a, lane, 1);
        float cst[32];
#pragma unroll
        for (int q = 0; q < 32; ++q) cst[q] = 0.f;

        for (int t = 0; t < T_LEN; ++t) {
#pragma unroll
            for (int jt = 0; jt < 4; ++jt) {
                v8f gi = wmma16(a1, load_bfrag(lds_w, lane, 1, jt),
                         wmma16(a0, load_bfrag(lds_w, lane, 0, jt), xg[jt]));
                v8f gf = wmma16(a1, load_bfrag(lds_w, lane, 1, 4 + jt),
                         wmma16(a0, load_bfrag(lds_w, lane, 0, 4 + jt), xg[4 + jt]));
                v8f gg = wmma16(a1, load_bfrag(lds_w, lane, 1, 8 + jt),
                         wmma16(a0, load_bfrag(lds_w, lane, 0, 8 + jt), xg[8 + jt]));
                v8f go = wmma16(a1, load_bfrag(lds_w, lane, 1, 12 + jt),
                         wmma16(a0, load_bfrag(lds_w, lane, 0, 12 + jt), xg[12 + jt]));
                int j = jt * 16 + ln;
#pragma unroll
                for (int v = 0; v < 8; ++v) {
                    float ii = fast_sig(gi[v]);
                    float ff = fast_sig(gf[v]);
                    float g2 = fast_tanh(gg[v]);
                    float oo = fast_sig(go[v]);
                    float cc = ff * cst[jt * 8 + v] + ii * g2;
                    cst[jt * 8 + v] = cc;
                    float hh = oo * fast_tanh(cc);
                    if (j < 50) lds_a[(rbase + v) * 128 + j] = (_Float16)hh;
                }
            }
            __syncthreads();
            a0 = load_afrag(lds_a, lane, 0);
            a1 = load_afrag(lds_a, lane, 1);
            // layer-2 input projection of the fresh h via the extra N-tile
            v8f pp = wmma16(a1, load_bfrag(lds_w, lane, 1, 16),
                     wmma16(a0, load_bfrag(lds_w, lane, 0, 16), zero8()));
            int tt = dir ? (T_LEN - 1 - t) : t;
            if (ln < 8) {
                _Float16* wp = ws + ((size_t)tt * BATCH + (row0 + rbase)) * 16 +
                               (dir ? 8 : 0) + ln;
#pragma unroll
                for (int v = 0; v < 8; ++v) wp[(size_t)v * 16] = (_Float16)pp[v];
            }
        }
        __syncthreads();
    }
}

// Layer-2 (H=1 per direction) scalar scans + channel sum. One thread per row.
__global__ __launch_bounds__(256) void lstm2_kernel(
    const _Float16* __restrict__ ws,
    const float* __restrict__ Whh2f,
    const float* __restrict__ Whh2b,
    float* __restrict__ out)
{
    int r = blockIdx.x * blockDim.x + threadIdx.x;
    if (r >= BATCH) return;
    float wf[4], wb[4];
#pragma unroll
    for (int g = 0; g < 4; ++g) { wf[g] = Whh2f[g]; wb[g] = Whh2b[g]; }

    float h = 0.f, c = 0.f;
    for (int t = 0; t < T_LEN; ++t) {
        const _Float16* z = ws + ((size_t)t * BATCH + r) * 16;
        float g0 = (float)z[0] + (float)z[8]  + h * wf[0];
        float g1 = (float)z[1] + (float)z[9]  + h * wf[1];
        float g2 = (float)z[2] + (float)z[10] + h * wf[2];
        float g3 = (float)z[3] + (float)z[11] + h * wf[3];
        float ii = fast_sig(g0), ff = fast_sig(g1);
        float gg = fast_tanh(g2), oo = fast_sig(g3);
        c = ff * c + ii * gg;
        h = oo * fast_tanh(c);
        out[(size_t)r * T_LEN + t] = h;
    }
    h = 0.f; c = 0.f;
    for (int s = 0; s < T_LEN; ++s) {
        int t = T_LEN - 1 - s;
        const _Float16* z = ws + ((size_t)t * BATCH + r) * 16;
        float g0 = (float)z[4] + (float)z[12] + h * wb[0];
        float g1 = (float)z[5] + (float)z[13] + h * wb[1];
        float g2 = (float)z[6] + (float)z[14] + h * wb[2];
        float g3 = (float)z[7] + (float)z[15] + h * wb[3];
        float ii = fast_sig(g0), ff = fast_sig(g1);
        float gg = fast_tanh(g2), oo = fast_sig(g3);
        c = ff * c + ii * gg;
        h = oo * fast_tanh(c);
        out[(size_t)r * T_LEN + t] += h;
    }
}

extern "C" void kernel_launch(void* const* d_in, const int* in_sizes, int n_in,
                              void* d_out, int out_size, void* d_ws, size_t ws_size,
                              hipStream_t stream) {
    const float* x     = (const float*)d_in[0];
    const float* Wih1f = (const float*)d_in[1];
    const float* Whh1f = (const float*)d_in[2];
    const float* bih1f = (const float*)d_in[3];
    const float* bhh1f = (const float*)d_in[4];
    const float* Wih1b = (const float*)d_in[5];
    const float* Whh1b = (const float*)d_in[6];
    const float* bih1b = (const float*)d_in[7];
    const float* bhh1b = (const float*)d_in[8];
    const float* Wih2f = (const float*)d_in[9];
    const float* Whh2f = (const float*)d_in[10];
    const float* bih2f = (const float*)d_in[11];
    const float* bhh2f = (const float*)d_in[12];
    const float* Wih2b = (const float*)d_in[13];
    const float* Whh2b = (const float*)d_in[14];
    const float* bih2b = (const float*)d_in[15];
    const float* bhh2b = (const float*)d_in[16];

    _Float16* ws = (_Float16*)d_ws;  // needs 512*1024*16*2 = 16 MB
    float* out = (float*)d_out;

    lstm1_kernel<<<BATCH / 16, 32, 0, stream>>>(
        x, Wih1f, Whh1f, bih1f, bhh1f, Wih1b, Whh1b, bih1b, bhh1b,
        Wih2f, bih2f, bhh2f, Wih2b, bih2b, bhh2b, ws);
    lstm2_kernel<<<BATCH / 256, 256, 0, stream>>>(ws, Whh2f, Whh2b, out);
}
